// TxModel_87402584473920
// MI455X (gfx1250) — matrix operand order repaired and checked by reading, not compiled, not run on hardware
//
#include <hip/hip_runtime.h>

// ---------------------------------------------------------------------------
// Fused tiny-transformer encoder for MI455X (gfx1250, wave32).
// One wave (32 threads) == one workgroup == one batch-element pipeline.
// All matmuls on v_wmma_f32_16x16x32_f16; softmax/LN/pool in VALU with
// wave32 shuffle reductions. x is read exactly once from HBM (205 MB floor).
// Round-2 changes: float4 x staging, per-head scores (<=4 live accs),
// max-free softmax, VGPR cap via launch_bounds to stay under 256 regs.
// ---------------------------------------------------------------------------

typedef __attribute__((ext_vector_type(8)))  _Float16 v8h;
typedef __attribute__((ext_vector_type(16))) _Float16 v16h;
typedef __attribute__((ext_vector_type(8)))  float    v8f;

#define B_TOTAL 16384
#define KSEQ    64
#define DIN     49
#define NBLK    4096
#define BPB     4          // batches per block (persistent loop)

__device__ __forceinline__ v8f wmma_f16(v16h a, v16h b, v8f c) {
  // D = A(16x32 f16) * B(32x16 f16) + C(16x16 f32)
  return __builtin_amdgcn_wmma_f32_16x16x32_f16(false, a, false, b,
                                                (short)0, c, false, false);
}

// A-fragment (16x32, f16) from row-major LDS buffer, row stride rs halves.
// ISA layout: lanes 0-15 row M, K chunks {0..7,16..23}; lanes 16-31 {8..15,24..31}.
__device__ __forceinline__ v16h ld_afrag(const _Float16* buf, int rs, int mrow, int kbase) {
  const int lane = threadIdx.x & 31;
  const _Float16* p = buf + (mrow + (lane & 15)) * rs + kbase + ((lane >> 4) << 3);
  v8h lo = *(const v8h*)(p);
  v8h hi = *(const v8h*)(p + 16);
  return __builtin_shufflevector(lo, hi, 0,1,2,3,4,5,6,7,8,9,10,11,12,13,14,15);
}

// B-fragment (32x16, f16): B[k][n] = buf[n][k] with buf row-major [N][K].
// ISA layout: lane holds column n=lane&15; lanes 0-15 K {0..7,16..23}, lanes 16-31 K {8..15,24..31}.
__device__ __forceinline__ v16h ld_bfrag(const _Float16* buf, int rs, int nrow, int kbase) {
  const int lane = threadIdx.x & 31;
  const _Float16* p = buf + (nrow + (lane & 15)) * rs + kbase + ((lane >> 4) << 3);
  v8h lo = *(const v8h*)(p);
  v8h hi = *(const v8h*)(p + 16);
  return __builtin_shufflevector(lo, hi, 0,1,2,3,4,5,6,7,8,9,10,11,12,13,14,15);
}

__global__ __launch_bounds__(32, 4)
void tx_fused_kernel(const float* __restrict__ x,
                     const float* __restrict__ proj_w, const float* __restrict__ proj_b,
                     const float* __restrict__ pos,
                     const float* __restrict__ in_proj_w, const float* __restrict__ in_proj_b,
                     const float* __restrict__ out_w, const float* __restrict__ out_b,
                     const float* __restrict__ ln1_g, const float* __restrict__ ln1_b,
                     const float* __restrict__ lin1_w, const float* __restrict__ lin1_b,
                     const float* __restrict__ lin2_w, const float* __restrict__ lin2_b,
                     const float* __restrict__ ln2_g, const float* __restrict__ ln2_b,
                     const float* __restrict__ head_w, const float* __restrict__ head_b,
                     float* __restrict__ outp) {
  // ---- LDS -----------------------------------------------------------------
  __shared__ _Float16 xsb[64 * 128];   // phase1: x tile [64][64]; phase2: probs [64][128]
  __shared__ _Float16 hbuf[64 * 32];   // h / LN1(h) (cols 16..31 stay zero)
  __shared__ _Float16 abuf[64 * 32];   // q -> ctx -> relu(ff1)
  __shared__ _Float16 kbuf[128 * 32];  // head-masked K^T: row kk=h*64+key, col dall
  __shared__ _Float16 vtb [16 * 128];  // head-masked V^T: row dall, col kk=h*64+key
  __shared__ _Float16 wp  [16 * 64];   // proj_w  [16][64pad]
  __shared__ _Float16 wqkv[48 * 32];   // in_proj [48][32pad]
  __shared__ _Float16 wo  [16 * 32];   // out_w   [16][32pad]
  __shared__ _Float16 w1  [32 * 32];   // lin1_w  [32][32pad]
  __shared__ _Float16 w2  [16 * 32];   // lin2_w  [16][32]
  __shared__ float    posb[64 * 16];   // positional emb (f32)
  __shared__ float    cb  [224];       // biases / LN params / head

  const int lane = threadIdx.x & 31;
  const int lrow = lane & 15;
  const int lhi  = lane >> 4;

  // ---- stage weights / constants (once per block) --------------------------
  for (int i = lane; i < 16 * 64;  i += 32) wp[i]   = (_Float16)0.f;
  for (int i = lane; i < 48 * 32;  i += 32) wqkv[i] = (_Float16)0.f;
  for (int i = lane; i < 16 * 32;  i += 32) wo[i]   = (_Float16)0.f;
  for (int i = lane; i < 32 * 32;  i += 32) w1[i]   = (_Float16)0.f;
  for (int i = lane; i < 128 * 32; i += 32) kbuf[i] = (_Float16)0.f;
  for (int i = lane; i < 16 * 128; i += 32) vtb[i]  = (_Float16)0.f;
  for (int i = lane; i < 64 * 32;  i += 32) hbuf[i] = (_Float16)0.f;
  for (int i = lane; i < 64 * 32;  i += 32) abuf[i] = (_Float16)0.f;

  for (int i = lane; i < 16 * DIN; i += 32) {           // proj_w [16][49] -> [16][64]
    int r = i / DIN, c = i - r * DIN;
    wp[r * 64 + c] = (_Float16)proj_w[i];
  }
  for (int i = lane; i < 48 * 16; i += 32) wqkv[(i >> 4) * 32 + (i & 15)] = (_Float16)in_proj_w[i];
  for (int i = lane; i < 16 * 16; i += 32) wo  [(i >> 4) * 32 + (i & 15)] = (_Float16)out_w[i];
  for (int i = lane; i < 32 * 16; i += 32) w1  [(i >> 4) * 32 + (i & 15)] = (_Float16)lin1_w[i];
  for (int i = lane; i < 16 * 32; i += 32) w2[i] = (_Float16)lin2_w[i];
  for (int i = lane; i < 64 * 16; i += 32) posb[i] = pos[i];

  if (lane < 16) {
    cb[lane]       = proj_b[lane];
    cb[64 + lane]  = out_b[lane];
    cb[112 + lane] = lin2_b[lane];
    cb[128 + lane] = ln1_g[lane];
    cb[144 + lane] = ln1_b[lane];
    cb[160 + lane] = ln2_g[lane];
    cb[176 + lane] = ln2_b[lane];
    cb[192 + lane] = head_w[lane];
  }
  if (lane < 32) cb[80 + lane] = lin1_b[lane];
  for (int i = lane; i < 48; i += 32) cb[16 + i] = in_proj_b[i];
  if (lane == 0) cb[208] = head_b[0];
  __syncthreads();

  const float INVS = 0.35355339059327f;  // 1/sqrt(HD=8)

  for (int bi = 0; bi < BPB; ++bi) {
    const int b = blockIdx.x * BPB + bi;
    const float4* xb4 = (const float4*)(x + (size_t)b * (KSEQ * DIN));  // 784 float4

    // prefetch next batch's x slab into cache hierarchy
    if (b + 1 < B_TOTAL) {
      const char* pn = (const char*)(x + (size_t)(b + 1) * (KSEQ * DIN));
      #pragma unroll
      for (int t = 0; t < 4; ++t) __builtin_prefetch(pn + (t * 32 + lane) * 128, 0, 0);
    }

    // ---- stage x tile: f32 -> f16, [64][49] -> [64][64pad] ----------------
    #pragma unroll
    for (int i = 0; i < 32; ++i)  // zero 4096 halves = 1024 u64
      ((unsigned long long*)xsb)[i * 32 + lane] = 0ULL;
    // re-zero abuf pad cols 16..31 (ff1 of previous batch wrote them)
    for (int i = lane; i < 1024; i += 32) abuf[(i >> 4) * 32 + 16 + (i & 15)] = (_Float16)0.f;
    #pragma unroll 1
    for (int t = 0; t < 25; ++t) {  // 784 float4, coalesced b128 loads
      int i4 = t * 32 + lane;
      if (i4 < 784) {
        float4 v = xb4[i4];
        int f   = i4 * 4;
        int row = f / DIN;
        int col = f - row * DIN;
        float e[4] = {v.x, v.y, v.z, v.w};
        #pragma unroll
        for (int k = 0; k < 4; ++k) {
          int c = col + k, r = row;
          if (c >= DIN) { c -= DIN; r += 1; }   // wraps at most once (4 < 49)
          xsb[r * 64 + c] = (_Float16)e[k];
        }
      }
    }
    __syncthreads();

    // ---- proj: h = x @ proj_w^T + b + pos  -> hbuf[64][16] ----------------
    #pragma unroll
    for (int mt = 0; mt < 4; ++mt) {
      v8f acc;
      #pragma unroll
      for (int r = 0; r < 8; ++r)
        acc[r] = cb[lrow] + posb[(mt * 16 + lhi * 8 + r) * 16 + lrow];
      acc = wmma_f16(ld_afrag(xsb, 64, mt * 16, 0),  ld_bfrag(wp, 64, 0, 0),  acc);
      acc = wmma_f16(ld_afrag(xsb, 64, mt * 16, 32), ld_bfrag(wp, 64, 0, 32), acc);
      #pragma unroll
      for (int r = 0; r < 8; ++r)
        hbuf[(mt * 16 + lhi * 8 + r) * 32 + lrow] = (_Float16)acc[r];
    }
    __syncthreads();

    // ---- qkv = h @ in_proj^T + b ------------------------------------------
    #pragma unroll
    for (int mt = 0; mt < 4; ++mt) {
      v16h ah = ld_afrag(hbuf, 32, mt * 16, 0);
      #pragma unroll
      for (int nt = 0; nt < 3; ++nt) {
        v8f acc;
        #pragma unroll
        for (int r = 0; r < 8; ++r) acc[r] = cb[16 + nt * 16 + lrow];
        acc = wmma_f16(ah, ld_bfrag(wqkv, 32, nt * 16, 0), acc);
        const int dall = lrow, hh = lrow >> 3;
        #pragma unroll
        for (int r = 0; r < 8; ++r) {
          const int row = mt * 16 + lhi * 8 + r;
          _Float16 val = (_Float16)acc[r];
          if (nt == 0)      abuf[row * 32 + lrow] = val;                       // Q
          else if (nt == 1) kbuf[((hh << 6) + row) * 32 + dall] = val;         // masked K^T
          else              vtb[dall * 128 + (hh << 6) + row] = val;           // masked V^T
        }
      }
    }
    __syncthreads();

    // ---- scores + softmax (per head: only 4 live accumulators) ------------
    // scores are bounded (|s|~O(1)) -> max-free softmax (shift-invariant).
    #pragma unroll
    for (int mt = 0; mt < 4; ++mt) {
      v16h aq = ld_afrag(abuf, 32, mt * 16, 0);
      #pragma unroll
      for (int h = 0; h < 2; ++h) {
        v8f sc[4];
        #pragma unroll
        for (int ntl = 0; ntl < 4; ++ntl) {
          v8f z;
          #pragma unroll
          for (int r = 0; r < 8; ++r) z[r] = 0.f;
          sc[ntl] = wmma_f16(aq, ld_bfrag(kbuf, 32, (h * 4 + ntl) * 16, 0), z);
        }
        #pragma unroll
        for (int r = 0; r < 8; ++r) {
          const int row = mt * 16 + lhi * 8 + r;
          float e0 = __expf(sc[0][r] * INVS), e1 = __expf(sc[1][r] * INVS);
          float e2 = __expf(sc[2][r] * INVS), e3 = __expf(sc[3][r] * INVS);
          float sm = (e0 + e1) + (e2 + e3);
          sm += __shfl_xor(sm, 1);
          sm += __shfl_xor(sm, 2);
          sm += __shfl_xor(sm, 4);
          sm += __shfl_xor(sm, 8);
          float rs_ = 1.0f / sm;
          _Float16* prow = xsb + row * 128 + h * 64 + lrow;
          prow[0]  = (_Float16)(e0 * rs_);
          prow[16] = (_Float16)(e1 * rs_);
          prow[32] = (_Float16)(e2 * rs_);
          prow[48] = (_Float16)(e3 * rs_);
        }
      }
    }
    __syncthreads();

    // ---- ctx = probs[64x128] @ maskedV^T  -> abuf[64][16] ------------------
    #pragma unroll
    for (int mt = 0; mt < 4; ++mt) {
      v8f acc;
      #pragma unroll
      for (int r = 0; r < 8; ++r) acc[r] = 0.f;
      #pragma unroll
      for (int ks = 0; ks < 4; ++ks)
        acc = wmma_f16(ld_afrag(xsb, 128, mt * 16, ks * 32),
                       ld_bfrag(vtb, 128, 0, ks * 32), acc);
      #pragma unroll
      for (int r = 0; r < 8; ++r)
        abuf[(mt * 16 + lhi * 8 + r) * 32 + lrow] = (_Float16)acc[r];
    }
    __syncthreads();

    // ---- out-proj + residual + LN1 -> hbuf --------------------------------
    #pragma unroll
    for (int mt = 0; mt < 4; ++mt) {
      v8f acc;
      #pragma unroll
      for (int r = 0; r < 8; ++r) acc[r] = cb[64 + lrow];
      acc = wmma_f16(ld_afrag(abuf, 32, mt * 16, 0), ld_bfrag(wo, 32, 0, 0), acc);
      #pragma unroll
      for (int r = 0; r < 8; ++r) {
        const int row = mt * 16 + lhi * 8 + r;
        float v = acc[r] + (float)hbuf[row * 32 + lrow];
        float s1 = v, s2 = v * v;
        #pragma unroll
        for (int off = 1; off < 16; off <<= 1) {
          s1 += __shfl_xor(s1, off);
          s2 += __shfl_xor(s2, off);
        }
        float mean = s1 * 0.0625f;
        float var  = s2 * 0.0625f - mean * mean;
        float y = (v - mean) * rsqrtf(var + 1e-5f) * cb[128 + lrow] + cb[144 + lrow];
        hbuf[row * 32 + lrow] = (_Float16)y;
      }
    }
    __syncthreads();

    // ---- ff1 + relu -> abuf[64][32] ---------------------------------------
    #pragma unroll
    for (int mt = 0; mt < 4; ++mt) {
      v16h ahf = ld_afrag(hbuf, 32, mt * 16, 0);
      #pragma unroll
      for (int nt = 0; nt < 2; ++nt) {
        v8f acc;
        #pragma unroll
        for (int r = 0; r < 8; ++r) acc[r] = cb[80 + nt * 16 + lrow];
        acc = wmma_f16(ahf, ld_bfrag(w1, 32, nt * 16, 0), acc);
        #pragma unroll
        for (int r = 0; r < 8; ++r)
          abuf[(mt * 16 + lhi * 8 + r) * 32 + nt * 16 + lrow] = (_Float16)fmaxf(acc[r], 0.f);
      }
    }
    __syncthreads();

    // ---- ff2 + residual + LN2 + mean-pool + head --------------------------
    float pool = 0.f;
    #pragma unroll
    for (int mt = 0; mt < 4; ++mt) {
      v8f acc;
      #pragma unroll
      for (int r = 0; r < 8; ++r) acc[r] = cb[112 + lrow];
      acc = wmma_f16(ld_afrag(abuf, 32, mt * 16, 0), ld_bfrag(w2, 32, 0, 0), acc);
      #pragma unroll
      for (int r = 0; r < 8; ++r) {
        const int row = mt * 16 + lhi * 8 + r;
        float v = acc[r] + (float)hbuf[row * 32 + lrow];
        float s1 = v, s2 = v * v;
        #pragma unroll
        for (int off = 1; off < 16; off <<= 1) {
          s1 += __shfl_xor(s1, off);
          s2 += __shfl_xor(s2, off);
        }
        float mean = s1 * 0.0625f;
        float var  = s2 * 0.0625f - mean * mean;
        float y = (v - mean) * rsqrtf(var + 1e-5f) * cb[160 + lrow] + cb[176 + lrow];
        pool += y;
      }
    }
    float pw = pool * cb[192 + lrow];
    pw += __shfl_xor(pw, 1);
    pw += __shfl_xor(pw, 2);
    pw += __shfl_xor(pw, 4);
    pw += __shfl_xor(pw, 8);
    pw += __shfl_xor(pw, 16);
    if (lane == 0) outp[b] = pw * (1.0f / 64.0f) + cb[208];
    __syncthreads();  // protect xsb/hbuf/abuf before next batch overwrites
  }
}

extern "C" void kernel_launch(void* const* d_in, const int* in_sizes, int n_in,
                              void* d_out, int out_size, void* d_ws, size_t ws_size,
                              hipStream_t stream) {
  (void)in_sizes; (void)n_in; (void)out_size; (void)d_ws; (void)ws_size;
  const float* x         = (const float*)d_in[0];
  const float* proj_w    = (const float*)d_in[1];
  const float* proj_b    = (const float*)d_in[2];
  const float* pos       = (const float*)d_in[3];
  const float* in_proj_w = (const float*)d_in[4];
  const float* in_proj_b = (const float*)d_in[5];
  const float* out_w     = (const float*)d_in[6];
  const float* out_b     = (const float*)d_in[7];
  const float* ln1_g     = (const float*)d_in[8];
  const float* ln1_b     = (const float*)d_in[9];
  const float* lin1_w    = (const float*)d_in[10];
  const float* lin1_b    = (const float*)d_in[11];
  const float* lin2_w    = (const float*)d_in[12];
  const float* lin2_b    = (const float*)d_in[13];
  const float* ln2_g     = (const float*)d_in[14];
  const float* ln2_b     = (const float*)d_in[15];
  const float* head_w    = (const float*)d_in[16];
  const float* head_b    = (const float*)d_in[17];

  hipLaunchKernelGGL(tx_fused_kernel, dim3(NBLK), dim3(32), 0, stream,
                     x, proj_w, proj_b, pos, in_proj_w, in_proj_b, out_w, out_b,
                     ln1_g, ln1_b, lin1_w, lin1_b, lin2_w, lin2_b, ln2_g, ln2_b,
                     head_w, head_b, (float*)d_out);
}
